// LambdaLayer_70394513981690
// MI455X (gfx1250) — matrix operand order repaired and checked by reading, not compiled, not run on hardware
//
#include <hip/hip_runtime.h>

typedef __attribute__((ext_vector_type(16))) __bf16 v16bf;
typedef __attribute__((ext_vector_type(8)))  float  v8f;
typedef __attribute__((ext_vector_type(2)))  float  v2f;

union Frag { v16bf v; unsigned u[8]; };

__device__ __forceinline__ unsigned short f2bf(float x) {
    __bf16 h = (__bf16)x;
    return __builtin_bit_cast(unsigned short, h);
}
__device__ __forceinline__ float bf2f(unsigned short u) {
    unsigned w = ((unsigned)u) << 16;
    return __builtin_bit_cast(float, w);
}

// --------------------------- async copy to LDS ------------------------------
#if defined(__gfx1250__) && __has_builtin(__builtin_amdgcn_global_load_async_to_lds_b128)
#define HAVE_ASYNC 1
typedef int vint4 __attribute__((vector_size(16)));
typedef __attribute__((address_space(1))) vint4 as1_vint4;   // global int4
typedef __attribute__((address_space(3))) vint4 as3_vint4;   // LDS int4
#else
#define HAVE_ASYNC 0
#endif

__device__ __forceinline__ void cp16(const void* g, void* l) {
#if HAVE_ASYNC
    __builtin_amdgcn_global_load_async_to_lds_b128((as1_vint4*)g, (as3_vint4*)l, 0, 0);
#else
    *(uint4*)l = *(const uint4*)g;
#endif
}

#if HAVE_ASYNC
#if __has_builtin(__builtin_amdgcn_s_wait_asynccnt)
#define WAIT_ASYNC(n) __builtin_amdgcn_s_wait_asynccnt(n)
#else
#define WAIT_ASYNC(n) asm volatile("s_wait_asynccnt %0" ::"n"(n) : "memory")
#endif
#else
#define WAIT_ASYNC(n)
#endif

// ---------------------------------------------------------------------------
// K1: keys = c @ Wk^T, valraw = c @ Wv^T  via V_WMMA_F32_16X16X4_F32
// grid (64 m-tiles, 32 b), 160 threads = 5 waves; wave w -> output cols [16w,16w+16)
// (w==0 -> 16 key cols; w=1..4 -> 64 value cols).  M=16 rows of c, K=256.
// A frag (16x4 f32): lane row = lane&15, v2f = row[4s + 2*half .. +1]
// B frag (4x16 f32): lane col = lane&15, v2f = Wcol[4s + 2*half .. +1]
// ---------------------------------------------------------------------------
__global__ __launch_bounds__(160) void proj_kv_kernel(
    const float* __restrict__ c, const float* __restrict__ Wk,
    const float* __restrict__ Wv, float* __restrict__ keys,
    float* __restrict__ valraw)
{
    __shared__ __align__(16) float ct[16 * 256];
    const int b = blockIdx.y, m0 = blockIdx.x * 16, tid = threadIdx.x;
    const float4* src = (const float4*)(c + ((size_t)b * 1024 + m0) * 256);
    float4* dst4 = (float4*)ct;
    for (int i = tid; i < 1024; i += 160) dst4[i] = src[i];
    __syncthreads();

    const int w = tid >> 5, lane = tid & 31;
    const int half = lane >> 4, lr = lane & 15;
    const float* Wrow = (w == 0) ? (Wk + (size_t)lr * 256)
                                 : (Wv + (size_t)((w - 1) * 16 + lr) * 256);
    const float* arow = ct + lr * 256;

    v8f acc = {};
    #pragma unroll 4
    for (int s = 0; s < 64; ++s) {
        int off = 4 * s + 2 * half;
        v2f a = *(const v2f*)(arow + off);
        v2f bb = *(const v2f*)(Wrow + off);
        acc = __builtin_amdgcn_wmma_f32_16x16x4_f32(false, a, false, bb,
                                                    (short)0, acc, false, false);
    }
    #pragma unroll
    for (int e = 0; e < 8; ++e) {
        int m = m0 + e + 8 * half;
        if (w == 0) keys[((size_t)b * 1024 + m) * 16 + lr] = acc[e];
        else        valraw[((size_t)b * 1024 + m) * 64 + (w - 1) * 16 + lr] = acc[e];
    }
}

// ---------------------------------------------------------------------------
// K2: qraw = xf @ Wq^T  (same WMMA-f32 scheme, 64 output cols -> 4 waves)
// ---------------------------------------------------------------------------
__global__ __launch_bounds__(128) void proj_q_kernel(
    const float* __restrict__ x, const float* __restrict__ Wq,
    float* __restrict__ qraw)
{
    __shared__ __align__(16) float xt[16 * 256];
    const int b = blockIdx.y, n0 = blockIdx.x * 16, tid = threadIdx.x;
    const float4* src = (const float4*)(x + ((size_t)b * 1024 + n0) * 256);
    float4* dst4 = (float4*)xt;
    for (int i = tid; i < 1024; i += 128) dst4[i] = src[i];
    __syncthreads();

    const int w = tid >> 5, lane = tid & 31;
    const int half = lane >> 4, lr = lane & 15;
    const float* Wrow = Wq + (size_t)(w * 16 + lr) * 256;
    const float* arow = xt + lr * 256;

    v8f acc = {};
    #pragma unroll 4
    for (int s = 0; s < 64; ++s) {
        int off = 4 * s + 2 * half;
        v2f a = *(const v2f*)(arow + off);
        v2f bb = *(const v2f*)(Wrow + off);
        acc = __builtin_amdgcn_wmma_f32_16x16x4_f32(false, a, false, bb,
                                                    (short)0, acc, false, false);
    }
    #pragma unroll
    for (int e = 0; e < 8; ++e) {
        int n = n0 + e + 8 * half;
        qraw[((size_t)b * 1024 + n) * 64 + w * 16 + lr] = acc[e];
    }
}

// ---------------------------------------------------------------------------
// K3: per-m BN stats of valraw over (b, v); fold gamma/beta into scale/shift
// ---------------------------------------------------------------------------
__global__ __launch_bounds__(256) void vstats_kernel(
    const float* __restrict__ valraw, const float* __restrict__ gamma,
    const float* __restrict__ beta, float* __restrict__ vscale,
    float* __restrict__ vshift)
{
    __shared__ float rs[256], rq[256];
    const int m = blockIdx.x, tid = threadIdx.x;
    float s = 0.f, s2 = 0.f;
    for (int idx = tid; idx < 2048; idx += 256) {
        int bb = idx >> 6, v = idx & 63;
        float xv = valraw[((size_t)bb * 1024 + m) * 64 + v];
        s += xv; s2 += xv * xv;
    }
    rs[tid] = s; rq[tid] = s2; __syncthreads();
    for (int st = 128; st > 0; st >>= 1) {
        if (tid < st) { rs[tid] += rs[tid + st]; rq[tid] += rq[tid + st]; }
        __syncthreads();
    }
    if (tid == 0) {
        float mean = rs[0] * (1.f / 2048.f);
        float var  = rq[0] * (1.f / 2048.f) - mean * mean;
        float sc   = rsqrtf(var + 1e-5f) * gamma[m];
        vscale[m] = sc;
        vshift[m] = beta[m] - mean * sc;
    }
}

// ---------------------------------------------------------------------------
// K4: per-k BN stats of qraw over (b, n, h)
// ---------------------------------------------------------------------------
__global__ __launch_bounds__(256) void qstats_kernel(
    const float* __restrict__ qraw, const float* __restrict__ gamma,
    const float* __restrict__ beta, float* __restrict__ qscale,
    float* __restrict__ qshift)
{
    __shared__ float rs[256], rq[256];
    const int k = blockIdx.x, tid = threadIdx.x;
    float s = 0.f, s2 = 0.f;
    for (int idx = tid; idx < 131072; idx += 256) {
        int bb = idx >> 12, rem = idx & 4095;
        int n = rem >> 2, hh = rem & 3;
        float xv = qraw[((size_t)bb * 1024 + n) * 64 + k * 4 + hh];
        s += xv; s2 += xv * xv;
    }
    rs[tid] = s; rq[tid] = s2; __syncthreads();
    for (int st = 128; st > 0; st >>= 1) {
        if (tid < st) { rs[tid] += rs[tid + st]; rq[tid] += rq[tid + st]; }
        __syncthreads();
    }
    if (tid == 0) {
        float mean = rs[0] * (1.f / 131072.f);
        float var  = rq[0] * (1.f / 131072.f) - mean * mean;
        float sc   = rsqrtf(var + 1e-5f) * gamma[k];
        qscale[k] = sc;
        qshift[k] = beta[k] - mean * sc;
    }
}

// ---------------------------------------------------------------------------
// K5: softmax over m per (b,k); stores transposed skT[b][k][m]
// ---------------------------------------------------------------------------
__global__ __launch_bounds__(256) void softmax_kernel(
    const float* __restrict__ keys, float* __restrict__ skT)
{
    __shared__ float red[256];
    const int k = blockIdx.x, b = blockIdx.y, tid = threadIdx.x;
    const float* kp = keys + (size_t)b * 16384 + k;
    float mx = -3.4e38f;
    for (int m = tid; m < 1024; m += 256) mx = fmaxf(mx, kp[(size_t)m * 16]);
    red[tid] = mx; __syncthreads();
    for (int st = 128; st > 0; st >>= 1) {
        if (tid < st) red[tid] = fmaxf(red[tid], red[tid + st]);
        __syncthreads();
    }
    float gmax = red[0]; __syncthreads();
    float* dst = skT + ((size_t)b * 16 + k) * 1024;
    float s = 0.f;
    for (int m = tid; m < 1024; m += 256) {
        float e = expf(kp[(size_t)m * 16] - gmax);
        dst[m] = e; s += e;
    }
    red[tid] = s; __syncthreads();
    for (int st = 128; st > 0; st >>= 1) {
        if (tid < st) red[tid] += red[tid + st];
        __syncthreads();
    }
    float inv = 1.f / red[0];
    for (int m = tid; m < 1024; m += 256) dst[m] *= inv;
}

// ---------------------------------------------------------------------------
// K6: BN-apply values + transpose + bf16:   Vt[b][v][m]
// ---------------------------------------------------------------------------
__global__ __launch_bounds__(256) void vt_kernel(
    const float* __restrict__ valraw, const float* __restrict__ vscale,
    const float* __restrict__ vshift, unsigned short* __restrict__ Vt)
{
    __shared__ float tile[64 * 65];
    const int m0 = blockIdx.x * 64, b = blockIdx.y, tid = threadIdx.x;
    for (int i = tid; i < 4096; i += 256) {
        int mm = i >> 6, v = i & 63;
        int m = m0 + mm;
        float xv = valraw[((size_t)b * 1024 + m) * 64 + v] * vscale[m] + vshift[m];
        tile[v * 65 + mm] = xv;
    }
    __syncthreads();
    for (int r = 0; r < 8; ++r) {
        int p = tid + 256 * r;               // < 2048
        int v = p >> 5, mp = p & 31;
        unsigned u = (unsigned)f2bf(tile[v * 65 + 2 * mp]) |
                     ((unsigned)f2bf(tile[v * 65 + 2 * mp + 1]) << 16);
        ((unsigned*)(Vt + ((size_t)b * 64 + v) * 1024 + m0))[mp] = u;
    }
}

// ---------------------------------------------------------------------------
// K7: BN-apply queries (elementwise)
// ---------------------------------------------------------------------------
__global__ __launch_bounds__(256) void bnq_kernel(
    const float* __restrict__ qraw, const float* __restrict__ qscale,
    const float* __restrict__ qshift, float* __restrict__ qbn)
{
    size_t idx = (size_t)blockIdx.x * 256 + threadIdx.x;
    int k = (int)((idx & 63) >> 2);
    qbn[idx] = qraw[idx] * qscale[k] + qshift[k];
}

// ---------------------------------------------------------------------------
// K8: E[n][m][k] f32  ->  Ebt[n][k][m] bf16  (transpose via LDS)
// ---------------------------------------------------------------------------
__global__ __launch_bounds__(256) void econv_kernel(
    const float* __restrict__ E, unsigned short* __restrict__ Ebt)
{
    __shared__ float tl[16 * 130];
    const int n = blockIdx.x, tid = threadIdx.x;
    for (int c = 0; c < 8; ++c) {
        __syncthreads();
        for (int i = tid; i < 2048; i += 256) {
            int mm = i >> 4, kk = i & 15;
            tl[kk * 130 + mm] = E[((size_t)n * 1024 + c * 128 + mm) * 16 + kk];
        }
        __syncthreads();
        for (int r = 0; r < 4; ++r) {
            int p = tid + 256 * r;           // < 1024
            int kk = p >> 6, mp = p & 63;
            unsigned u = (unsigned)f2bf(tl[kk * 130 + 2 * mp]) |
                         ((unsigned)f2bf(tl[kk * 130 + 2 * mp + 1]) << 16);
            ((unsigned*)(Ebt + ((size_t)n * 16 + kk) * 1024 + c * 128))[mp] = u;
        }
    }
}

// ---------------------------------------------------------------------------
// K9: content_lambda[b][k][v] = sum_m skT[b][k][m] * Vt[b][v][m]
// ---------------------------------------------------------------------------
__global__ __launch_bounds__(256) void cl_kernel(
    const float* __restrict__ skT, const unsigned short* __restrict__ Vt,
    float* __restrict__ cl)
{
    const int b = blockIdx.x, tid = threadIdx.x;
    for (int r = 0; r < 4; ++r) {
        int o = tid + 256 * r;               // < 1024
        int k = o >> 6, v = o & 63;
        const float* sk = skT + ((size_t)b * 16 + k) * 1024;
        const unsigned short* vr = Vt + ((size_t)b * 64 + v) * 1024;
        float s = 0.f;
        for (int m = 0; m < 1024; ++m) s += sk[m] * bf2f(vr[m]);
        cl[((size_t)b * 16 + k) * 64 + v] = s;
    }
}

// ---------------------------------------------------------------------------
// MAIN: per (b, 16-n tile):
//   PL[n][k][v] = sum_m Ebt[n][k][m] * Vt[b][v][m]      (bf16 WMMA, f32 acc)
//   out[b][h*64+v][n] = sum_k qbn[b][n][k*4+h] * (PL[n][k][v] + cl[b][k][v])
// 128 threads = 4 waves; wave w owns nl in [4w,4w+4), all 4 v-tiles.
// Double-buffered LDS (chunk = 64 m), async global->LDS when available.
// ---------------------------------------------------------------------------
#define CH 64
#define NCHUNK 16
#define ASYNC_PER_FILL 20   // per-thread b128 issues per fill (16 E + 4 V)

__global__ __launch_bounds__(128) void lambda_main_kernel(
    const unsigned short* __restrict__ Ebt,   // [1024][16][1024] bf16
    const unsigned short* __restrict__ Vt,    // [32][64][1024] bf16
    const float* __restrict__ qbn,            // [32][1024][64]
    const float* __restrict__ cl,             // [32][16][64]
    float* __restrict__ out)                  // [32][256][1024]
{
    __shared__ __align__(16) unsigned short Ebuf[2 * 16 * 16 * CH]; // 64 KB
    __shared__ __align__(16) unsigned short Vbuf[2 * 64 * CH];      // 16 KB
    __shared__ float qs[16 * 64];                                   // 4 KB
    __shared__ float cls[16 * 64];                                  // 4 KB

    const int b   = blockIdx.y;
    const int n0  = blockIdx.x * 16;
    const int tid = threadIdx.x;
    const int lane = tid & 31;
    const int w    = tid >> 5;
    const int half = lane >> 4;
    const int lr   = lane & 15;

    // stage q tile and content_lambda
    for (int i = tid; i < 16 * 64; i += 128) {
        qs[i]  = qbn[((size_t)b * 1024 + n0 + (i >> 6)) * 64 + (i & 63)];
        cls[i] = cl[((size_t)b * 16) * 64 + i];
    }

    v8f acc[4][4];
    #pragma unroll
    for (int i = 0; i < 4; ++i)
        #pragma unroll
        for (int j = 0; j < 4; ++j) acc[i][j] = {};

    // fill of chunk 0 into stage 0
    {
        unsigned short* eb = Ebuf;
        unsigned short* vb = Vbuf;
        for (int idx = tid; idx < 2048; idx += 128) {
            int r = idx >> 3, q4 = idx & 7;
            int nl = r >> 4, k = r & 15;
            const uint4* g = (const uint4*)(Ebt + ((size_t)(n0 + nl) * 16 + k) * 1024) + q4;
            cp16(g, (uint4*)eb + r * 8 + q4);
        }
        for (int idx = tid; idx < 512; idx += 128) {
            int r = idx >> 3, q4 = idx & 7;
            const uint4* g = (const uint4*)(Vt + ((size_t)b * 64 + r) * 1024) + q4;
            cp16(g, (uint4*)vb + r * 8 + q4);
        }
    }

    for (int c = 0; c < NCHUNK; ++c) {
        const int stg = c & 1;
        if (c + 1 < NCHUNK) {
            unsigned short* eb = Ebuf + ((c + 1) & 1) * (16 * 16 * CH);
            unsigned short* vb = Vbuf + ((c + 1) & 1) * (64 * CH);
            for (int idx = tid; idx < 2048; idx += 128) {
                int r = idx >> 3, q4 = idx & 7;
                int nl = r >> 4, k = r & 15;
                const uint4* g = (const uint4*)(Ebt + ((size_t)(n0 + nl) * 16 + k) * 1024 +
                                                (size_t)(c + 1) * CH) + q4;
                cp16(g, (uint4*)eb + r * 8 + q4);
            }
            for (int idx = tid; idx < 512; idx += 128) {
                int r = idx >> 3, q4 = idx & 7;
                const uint4* g = (const uint4*)(Vt + ((size_t)b * 64 + r) * 1024 +
                                                (size_t)(c + 1) * CH) + q4;
                cp16(g, (uint4*)vb + r * 8 + q4);
            }
#if HAVE_ASYNC
            WAIT_ASYNC(ASYNC_PER_FILL);   // previous chunk's fills are done
#endif
        } else {
            WAIT_ASYNC(0);
        }
        __syncthreads();                   // stage `stg` ready for everyone

        const unsigned short* eb = Ebuf + stg * (16 * 16 * CH);
        const unsigned short* vb = Vbuf + stg * (64 * CH);
        for (int kk = 0; kk < 2; ++kk) {   // K = 2 x 32 per chunk
            Frag A[4], B[4];
            #pragma unroll
            for (int i = 0; i < 4; ++i) {
                int nl = w * 4 + i;
                const unsigned* base =
                    (const unsigned*)(eb + (nl * 16 + lr) * CH + kk * 32);
                #pragma unroll
                for (int j = 0; j < 8; ++j) {
                    int Kb = (j < 4 ? 2 * j : 16 + 2 * (j - 4)) + half * 8;
                    A[i].u[j] = base[Kb >> 1];
                }
            }
            #pragma unroll
            for (int j4 = 0; j4 < 4; ++j4) {
                int v = j4 * 16 + lr;
                const unsigned* base = (const unsigned*)(vb + v * CH + kk * 32);
                #pragma unroll
                for (int j = 0; j < 8; ++j) {
                    int Kb = (j < 4 ? 2 * j : 16 + 2 * (j - 4)) + half * 8;
                    B[j4].u[j] = base[Kb >> 1];
                }
            }
            #pragma unroll
            for (int i = 0; i < 4; ++i)
                #pragma unroll
                for (int j = 0; j < 4; ++j)
                    acc[i][j] = __builtin_amdgcn_wmma_f32_16x16x32_bf16(
                        false, A[i].v, false, B[j].v, (short)0, acc[i][j],
                        false, false);
        }
        __syncthreads();                   // done reading stage before refill
    }

    // dump PL into LDS (alias Ebuf: 16*16*64 f32 = 64 KB)
    float* Lbuf = (float*)Ebuf;
    #pragma unroll
    for (int i = 0; i < 4; ++i) {
        int nl = w * 4 + i;
        #pragma unroll
        for (int j = 0; j < 4; ++j) {
            int v = j * 16 + lr;
            #pragma unroll
            for (int e = 0; e < 8; ++e) {
                int k = e + half * 8;
                Lbuf[(nl * 16 + k) * 64 + v] = acc[i][j][e];
            }
        }
    }
    __syncthreads();

    // epilogue: fuse content lambda + contract with queries, coalesced over n
    for (int r = 0; r < 32; ++r) {
        int nl  = tid & 15;
        int dch = (tid >> 4) + r * 8;        // 0..255 output channel = h*64+v
        int h = dch >> 6, v = dch & 63;
        float s = 0.f;
        #pragma unroll
        for (int k = 0; k < 16; ++k)
            s += qs[nl * 64 + k * 4 + h] * (Lbuf[(nl * 16 + k) * 64 + v] + cls[k * 64 + v]);
        out[(size_t)b * 262144 + (size_t)dch * 1024 + n0 + nl] = s;
    }
}

// ---------------------------------------------------------------------------
extern "C" void kernel_launch(void* const* d_in, const int* in_sizes, int n_in,
                              void* d_out, int out_size, void* d_ws, size_t ws_size,
                              hipStream_t stream)
{
    (void)in_sizes; (void)n_in; (void)out_size; (void)ws_size;
    const float* x  = (const float*)d_in[0];
    const float* c  = (const float*)d_in[1];
    const float* Wq = (const float*)d_in[2];
    const float* Wk = (const float*)d_in[3];
    const float* Wv = (const float*)d_in[4];
    const float* E  = (const float*)d_in[5];
    const float* gv = (const float*)d_in[6];
    const float* bv = (const float*)d_in[7];
    const float* gq = (const float*)d_in[8];
    const float* bq = (const float*)d_in[9];
    float* out = (float*)d_out;

    char* ws = (char*)d_ws;
    size_t cur = 0;
    auto alloc = [&](size_t bytes) -> char* {
        char* p = ws + cur;
        cur = (cur + bytes + 255) & ~(size_t)255;
        return p;
    };
    float* keys     = (float*)alloc(32ull * 1024 * 16 * 4);   // 2 MB
    float* valraw   = (float*)alloc(32ull * 1024 * 64 * 4);   // 8 MB
    float* qraw     = (float*)alloc(32ull * 1024 * 64 * 4);   // 8 MB
    float* qbn      = (float*)alloc(32ull * 1024 * 64 * 4);   // 8 MB
    float* skT      = (float*)alloc(32ull * 16 * 1024 * 4);   // 2 MB
    float* clbuf    = (float*)alloc(32ull * 16 * 64 * 4);     // 128 KB
    float* vscale   = (float*)alloc(1024 * 4);
    float* vshift   = (float*)alloc(1024 * 4);
    float* qscale   = (float*)alloc(64);
    float* qshift   = (float*)alloc(64);
    unsigned short* Vt  = (unsigned short*)alloc(32ull * 64 * 1024 * 2);    // 4 MB
    unsigned short* Ebt = (unsigned short*)alloc(1024ull * 16 * 1024 * 2);  // 32 MB

    proj_kv_kernel<<<dim3(64, 32), 160, 0, stream>>>(c, Wk, Wv, keys, valraw);
    proj_q_kernel<<<dim3(64, 32), 128, 0, stream>>>(x, Wq, qraw);
    vstats_kernel<<<1024, 256, 0, stream>>>(valraw, gv, bv, vscale, vshift);
    qstats_kernel<<<16, 256, 0, stream>>>(qraw, gq, bq, qscale, qshift);
    softmax_kernel<<<dim3(16, 32), 256, 0, stream>>>(keys, skT);
    vt_kernel<<<dim3(16, 32), 256, 0, stream>>>(valraw, vscale, vshift, Vt);
    bnq_kernel<<<8192, 256, 0, stream>>>(qraw, qscale, qshift, qbn);
    econv_kernel<<<1024, 256, 0, stream>>>(E, Ebt);
    cl_kernel<<<32, 256, 0, stream>>>(skT, Vt, clbuf);
    lambda_main_kernel<<<dim3(64, 32), 128, 0, stream>>>(Ebt, Vt, qbn, clbuf, out);
}
